// Encoder_71244917506126
// MI455X (gfx1250) — matrix operand order repaired and checked by reading, not compiled, not run on hardware
//
#include <hip/hip_runtime.h>
#include <hip/hip_bf16.h>
#include <stdint.h>

// ---------------------------------------------------------------------------
// Problem constants (from the reference)
// ---------------------------------------------------------------------------
constexpr int L   = 4;
constexpr int S   = 2048;
constexpr int D   = 1024;
constexpr int DFF = 2048;
constexpr int H   = 8;
constexpr int DK  = 128;     // D / H
constexpr float NEGBIG = -1.0e9f;

typedef __attribute__((ext_vector_type(16))) _Float16 v16h;
typedef __attribute__((ext_vector_type(8)))  float    v8f;

union Frag {
    v16h     v;
    uint32_t u[8];
    _Float16 h[16];
};

// ---------------------------------------------------------------------------
// CDNA5 async Global->LDS copy (16B per lane), tracked by ASYNCcnt.
// LDS byte offset = low 32 bits of the generic pointer (ISA: flat LDS addr
// truncates to addr[31:0] = wave-relative LDS offset).
// ---------------------------------------------------------------------------
__device__ __forceinline__ void async_copy_b128(void* lds, const _Float16* g)
{
    uint32_t lds_off = (uint32_t)(uintptr_t)lds;
    asm volatile("global_load_async_to_lds_b128 %0, %1, off"
                 :: "v"(lds_off), "v"(g) : "memory");
}
__device__ __forceinline__ void wait_async_le3()
{
    asm volatile("s_wait_asynccnt 3" ::: "memory");
}
__device__ __forceinline__ void wait_async_le4()
{
    asm volatile("s_wait_asynccnt 4" ::: "memory");
}
__device__ __forceinline__ void wait_async_0()
{
    asm volatile("s_wait_asynccnt 0" ::: "memory");
}

// ---------------------------------------------------------------------------
// Weight f32 -> f16 with transpose: in [L][K][N] -> out [L][N][K]
// One-time pass; lets GEMM B tiles be staged with contiguous async b128 copies.
// ---------------------------------------------------------------------------
__global__ __launch_bounds__(256) void convert_transpose_f16(
    const float* __restrict__ in, _Float16* __restrict__ out, int K, int N)
{
    size_t i = (size_t)blockIdx.x * 256 + threadIdx.x;
    size_t per = (size_t)K * N;
    if (i >= (size_t)L * per) return;
    size_t l = i / per;
    size_t r = i % per;
    int k = (int)(r / N);
    int n = (int)(r % N);
    out[l * per + (size_t)n * K + k] = (_Float16)in[i];
}

// ---------------------------------------------------------------------------
// x = 2*enc_inputs + sinusoidal_pe ; dual f32/f16 write
// ---------------------------------------------------------------------------
__global__ __launch_bounds__(256) void pe_init_kernel(
    const float* __restrict__ in, float* __restrict__ x32, _Float16* __restrict__ x16)
{
    int i = blockIdx.x * 256 + threadIdx.x;
    if (i >= S * D) return;
    int s  = i / D;
    int d  = i % D;
    int i2 = d >> 1;
    float dv  = __expf((float)(2 * i2) * (-9.210340371976184f / (float)D)); // ln(10000)
    float ang = (float)s * dv;
    float pe  = (d & 1) ? __cosf(ang) : __sinf(ang);
    float v   = 2.0f * in[i] + pe;
    x32[i] = v;
    x16[i] = (_Float16)v;
}

// ---------------------------------------------------------------------------
// mvec: 1 = local token, -1 = global token, 0 = pad
// ---------------------------------------------------------------------------
__global__ void mvec_init_kernel(float* __restrict__ mvec, const int* __restrict__ Tp)
{
    int i = blockIdx.x * 256 + threadIdx.x;
    if (i < S) mvec[i] = (i < Tp[0]) ? 1.0f : 0.0f;
}
__global__ void mvec_global_kernel(float* __restrict__ mvec, const int* __restrict__ gidx)
{
    int i = threadIdx.x;
    if (i < 32) {
        int g = gidx[i] / 15;
        if (g >= 0 && g < S) mvec[g] = -1.0f;
    }
}

// ---------------------------------------------------------------------------
// WMMA GEMM: C(MxN) = A(MxK,f16,row) @ B(KxN)  with B given TRANSPOSED: BT[N][K].
// Block tile 128x64, 8 waves (one 16-row slab each, 4 N sub-tiles), BK=32.
// Double-buffered LDS, async Global->LDS staging (3 async ops/thread/stage).
// modes: 0 = f16 | 1 = f16 transposed out[n*ldT+m] | 2 = f32 acc+res | 3 = f16 ReLU
// ---------------------------------------------------------------------------
#define BM 128
#define BN 64
#define BK 32

__global__ __launch_bounds__(256) void gemm_f16_wmma(
    const _Float16* __restrict__ A, const _Float16* __restrict__ BT,
    const float* __restrict__ res, void* __restrict__ out,
    int M, int N, int K, int mode, int ldT)
{
    __shared__ __align__(16) _Float16 As[2][BM * 40];  // [buf][row][32 halves +8 pad]
    __shared__ __align__(16) _Float16 Bs[2][BN * 40];  // [buf][col n][32 halves +8 pad]

    const int tid  = threadIdx.x;
    const int lane = tid & 31;
    const int w    = tid >> 5;
    const int m0   = blockIdx.y * BM;
    const int n0   = blockIdx.x * BN;
    const int half = lane >> 4;
    const int nlo  = lane & 15;

    // staging maps (uniform per wave)
    const int ar = tid >> 1;                 // 0..127 A row
    const int ac = (tid & 1) * 16;           // 0/16   A half-group
    const int bn = tid >> 2;                 // 0..63  B(T) row (= output col n)
    const int bc = (tid & 3) * 8;            // 0/8/16/24 k chunk

    v8f acc[4] = {};

    auto stage = [&](int buf, int k0) {
        const _Float16* ga = A + (size_t)(m0 + ar) * K + k0 + ac;
        async_copy_b128(&As[buf][ar * 40 + ac],     ga);
        async_copy_b128(&As[buf][ar * 40 + ac + 8], ga + 8);
        const _Float16* gb = BT + (size_t)(n0 + bn) * K + k0 + bc;
        async_copy_b128(&Bs[buf][bn * 40 + bc],     gb);
    };

    stage(0, 0);
    int b = 0;
    for (int k0 = 0; k0 < K; k0 += BK) {
        const bool more = (k0 + BK) < K;
        if (more) stage(b ^ 1, k0 + BK);
        if (more) wait_async_le3(); else wait_async_0();
        __syncthreads();

        // A fragment (16x32 doc layout)
        Frag af;
        {
            const int kb = half * 8;
            const uint32_t* arow = (const uint32_t*)&As[b][(w * 16 + nlo) * 40];
            #pragma unroll
            for (int p = 0; p < 4; ++p) {
                af.u[p]     = arow[(kb >> 1) + p];
                af.u[4 + p] = arow[8 + (kb >> 1) + p];
            }
        }
        #pragma unroll
        for (int t = 0; t < 4; ++t) {
            Frag bf;
            const uint32_t* brow = (const uint32_t*)&Bs[b][(t * 16 + nlo) * 40];
            #pragma unroll
            for (int p = 0; p < 8; ++p) bf.u[p] = brow[(p >> 2) * 8 + half * 4 + (p & 3)];
            acc[t] = __builtin_amdgcn_wmma_f32_16x16x32_f16(
                false, af.v, false, bf.v, (short)0, acc[t], false, false);
        }
        __syncthreads();
        b ^= 1;
    }

    // epilogue (C layout: VGPR j -> row j / j+8 per lane half, lanes -> N)
    #pragma unroll
    for (int t = 0; t < 4; ++t) {
        #pragma unroll
        for (int j = 0; j < 8; ++j) {
            const int gm = m0 + w * 16 + j + 8 * half;
            const int gn = n0 + t * 16 + nlo;
            const float v = acc[t][j];
            const size_t idx = (size_t)gm * N + gn;
            if (mode == 0) {
                ((_Float16*)out)[idx] = (_Float16)v;
            } else if (mode == 1) {
                ((_Float16*)out)[(size_t)gn * ldT + gm] = (_Float16)v;
            } else if (mode == 2) {
                ((float*)out)[idx] = v + res[idx];
            } else {
                ((_Float16*)out)[idx] = (_Float16)(v > 0.0f ? v : 0.0f);
            }
        }
    }
}

// ---------------------------------------------------------------------------
// Flash-style local/global attention.
// Block = (head, 8 consecutive query tiles); 8 waves share double-buffered,
// async-staged K (32x128) and V^T (128x32) LDS slabs (4 async ops/thread/stage).
// Q,K: f16 [S][D] (head slab at col h*DK). VT: f16 [D][S]. ctx: f16 [S][D].
// ---------------------------------------------------------------------------
__global__ __launch_bounds__(256) void attn_wmma(
    const _Float16* __restrict__ Q, const _Float16* __restrict__ Km,
    const _Float16* __restrict__ VT, const float* __restrict__ mvec,
    _Float16* __restrict__ ctx)
{
    __shared__ __align__(16) _Float16 Ks[2][32 * 136];  // [key][dk 128 +8 pad]
    __shared__ __align__(16) _Float16 Vs[2][128 * 40];  // [dk][key 32 +8 pad]
    __shared__ __align__(16) _Float16 P[8 * 16 * 40];   // per-wave prob tiles

    const int tid   = threadIdx.x;
    const int lane  = tid & 31;
    const int w     = tid >> 5;
    const int h     = blockIdx.x >> 4;              // head
    const int qg    = blockIdx.x & 15;              // query-tile group
    const int qbase = (qg * 8 + w) * 16;
    const int half  = lane >> 4;
    const int nlo   = lane & 15;
    const float scale = 0.08838834764831845f;       // 1/sqrt(128)

    // staging maps (uniform per wave)
    const int sk_key = tid >> 3;                    // 0..31
    const int sk_c   = (tid & 7) * 16;              // dk chunk pairs (halves)
    const int sv_d   = tid >> 1;                    // 0..127
    const int sv_c   = (tid & 1) * 16;              // key chunk pairs (halves)

    auto stage = [&](int buf, int kt) {
        const int kb0 = kt * 32;
        const _Float16* gk = Km + (size_t)(kb0 + sk_key) * D + h * DK + sk_c;
        async_copy_b128(&Ks[buf][sk_key * 136 + sk_c],     gk);
        async_copy_b128(&Ks[buf][sk_key * 136 + sk_c + 8], gk + 8);
        const _Float16* gv = VT + (size_t)(h * DK + sv_d) * S + kb0 + sv_c;
        async_copy_b128(&Vs[buf][sv_d * 40 + sv_c],        gv);
        async_copy_b128(&Vs[buf][sv_d * 40 + sv_c + 8],    gv + 8);
    };

    // ---- cache Q fragments (4 dk-steps of 32) ----
    Frag qf[4];
    {
        const _Float16* qrow = Q + (size_t)(qbase + nlo) * D + h * DK;
        const int kb = half * 8;
        #pragma unroll
        for (int s2 = 0; s2 < 4; ++s2) {
            const uint32_t* p32 = (const uint32_t*)(qrow + s2 * 32);
            #pragma unroll
            for (int p = 0; p < 4; ++p) {
                qf[s2].u[p]     = p32[half * 8 + p];
                qf[s2].u[4 + p] = p32[half * 8 + 4 + p];
            }
        }
    }

    float mq[8];
    #pragma unroll
    for (int j = 0; j < 8; ++j) mq[j] = mvec[qbase + j + 8 * half];

    v8f  o[8] = {};
    float rm[8], rl[8];
    #pragma unroll
    for (int j = 0; j < 8; ++j) { rm[j] = -3.0e38f; rl[j] = 0.0f; }

    stage(0, 0);
    int b = 0;
    for (int kt = 0; kt < S / 32; ++kt) {
        const int kb0 = kt * 32;
        const bool more = (kt + 1) < (S / 32);
        if (more) stage(b ^ 1, kt + 1);
        if (more) wait_async_le4(); else wait_async_0();
        __syncthreads();

        // ---- scores = Q @ K^T over dk=128 in 4 steps ----
        v8f s0 = {}, s1 = {};
        #pragma unroll
        for (int s2 = 0; s2 < 4; ++s2) {
            Frag kf0, kf1;
            const uint32_t* k0p = (const uint32_t*)&Ks[b][nlo * 136 + s2 * 32 + half * 16];
            const uint32_t* k1p = (const uint32_t*)&Ks[b][(16 + nlo) * 136 + s2 * 32 + half * 16];
            #pragma unroll
            for (int p = 0; p < 8; ++p) { kf0.u[p] = k0p[p]; kf1.u[p] = k1p[p]; }
            s0 = __builtin_amdgcn_wmma_f32_16x16x32_f16(false, qf[s2].v, false, kf0.v, (short)0, s0, false, false);
            s1 = __builtin_amdgcn_wmma_f32_16x16x32_f16(false, qf[s2].v, false, kf1.v, (short)0, s1, false, false);
        }

        // ---- mask + online softmax (row reductions within 16-lane halves) ----
        const float mk0 = mvec[kb0 + nlo];
        const float mk1 = mvec[kb0 + 16 + nlo];
        const int key0 = kb0 + nlo;
        const int key1 = key0 + 16;
        float corr[8];
        #pragma unroll
        for (int j = 0; j < 8; ++j) {
            const int q = qbase + j + 8 * half;
            const float mqj = mq[j];
            int d0 = q - key0; if (d0 < 0) d0 = -d0;
            int d1 = q - key1; if (d1 < 0) d1 = -d1;
            const bool a0 = (mqj != 0.0f) && (mk0 != 0.0f) && (d0 <= 32 || mqj == -1.0f || mk0 == -1.0f);
            const bool a1 = (mqj != 0.0f) && (mk1 != 0.0f) && (d1 <= 32 || mqj == -1.0f || mk1 == -1.0f);
            const float v0 = a0 ? s0[j] * scale : NEGBIG;
            const float v1 = a1 ? s1[j] * scale : NEGBIG;
            float mt = fmaxf(v0, v1);
            mt = fmaxf(mt, __shfl_xor(mt, 1, 32));
            mt = fmaxf(mt, __shfl_xor(mt, 2, 32));
            mt = fmaxf(mt, __shfl_xor(mt, 4, 32));
            mt = fmaxf(mt, __shfl_xor(mt, 8, 32));
            const float mnew = fmaxf(rm[j], mt);
            const float c  = __expf(rm[j] - mnew);
            const float p0 = __expf(v0 - mnew);
            const float p1 = __expf(v1 - mnew);
            float rs = p0 + p1;
            rs += __shfl_xor(rs, 1, 32);
            rs += __shfl_xor(rs, 2, 32);
            rs += __shfl_xor(rs, 4, 32);
            rs += __shfl_xor(rs, 8, 32);
            rl[j] = rl[j] * c + rs;
            rm[j] = mnew;
            corr[j] = c;
            const int m = j + 8 * half;
            P[(w * 16 + m) * 40 + nlo]      = (_Float16)p0;
            P[(w * 16 + m) * 40 + 16 + nlo] = (_Float16)p1;
        }
        #pragma unroll
        for (int t = 0; t < 8; ++t)
            #pragma unroll
            for (int j = 0; j < 8; ++j) o[t][j] *= corr[j];

        asm volatile("s_wait_dscnt 0" ::: "memory");

        // ---- probs A-fragment from LDS ----
        Frag pf;
        {
            const int kb = half * 8;
            const uint32_t* prow = (const uint32_t*)&P[(w * 16 + nlo) * 40];
            #pragma unroll
            for (int p = 0; p < 4; ++p) {
                pf.u[p]     = prow[(kb >> 1) + p];
                pf.u[4 + p] = prow[8 + (kb >> 1) + p];
            }
        }
        // ---- O += P @ V (8 dk tiles) ----
        #pragma unroll
        for (int t = 0; t < 8; ++t) {
            Frag vf;
            const uint32_t* vrow = (const uint32_t*)&Vs[b][(t * 16 + nlo) * 40];
            #pragma unroll
            for (int p = 0; p < 8; ++p) vf.u[p] = vrow[(p >> 2) * 8 + half * 4 + (p & 3)];
            o[t] = __builtin_amdgcn_wmma_f32_16x16x32_f16(false, pf.v, false, vf.v, (short)0, o[t], false, false);
        }
        __syncthreads();
        b ^= 1;
    }

    // ---- normalize + store ctx ----
    #pragma unroll
    for (int j = 0; j < 8; ++j) {
        const float inv = 1.0f / rl[j];
        const int q = qbase + j + 8 * half;
        #pragma unroll
        for (int t = 0; t < 8; ++t)
            ctx[(size_t)q * D + h * DK + t * 16 + nlo] = (_Float16)(o[t][j] * inv);
    }
}

// ---------------------------------------------------------------------------
// LayerNorm over D=1024, one row per block; dual f32/f16 output
// ---------------------------------------------------------------------------
__global__ __launch_bounds__(256) void layernorm_kernel(
    const float* __restrict__ in, const float* __restrict__ g, const float* __restrict__ b,
    float* __restrict__ out32, _Float16* __restrict__ out16)
{
    __shared__ float red[256];
    const int row = blockIdx.x;
    const int tid = threadIdx.x;
    const float* x = in + (size_t)row * D;

    float s = 0.0f;
    for (int i = tid; i < D; i += 256) s += x[i];
    red[tid] = s; __syncthreads();
    for (int off = 128; off > 0; off >>= 1) {
        if (tid < off) red[tid] += red[tid + off];
        __syncthreads();
    }
    const float mu = red[0] / (float)D;
    __syncthreads();

    float v = 0.0f;
    for (int i = tid; i < D; i += 256) { float d = x[i] - mu; v += d * d; }
    red[tid] = v; __syncthreads();
    for (int off = 128; off > 0; off >>= 1) {
        if (tid < off) red[tid] += red[tid + off];
        __syncthreads();
    }
    const float rstd = rsqrtf(red[0] / (float)D + 1e-5f);

    for (int i = tid; i < D; i += 256) {
        const float y = g[i] * (x[i] - mu) * rstd + b[i];
        out32[(size_t)row * D + i] = y;
        out16[(size_t)row * D + i] = (_Float16)y;
    }
}

// ---------------------------------------------------------------------------
// Host-side orchestration
// ---------------------------------------------------------------------------
extern "C" void kernel_launch(void* const* d_in, const int* in_sizes, int n_in,
                              void* d_out, int out_size, void* d_ws, size_t ws_size,
                              hipStream_t stream)
{
    (void)in_sizes; (void)n_in; (void)out_size; (void)ws_size;

    const int*   Tp    = (const int*)  d_in[0];
    const float* enc   = (const float*)d_in[1];
    const int*   gidx  = (const int*)  d_in[2];
    const float* Wq    = (const float*)d_in[3];
    const float* Wk    = (const float*)d_in[4];
    const float* Wv    = (const float*)d_in[5];
    const float* Wo    = (const float*)d_in[6];
    const float* ln1g  = (const float*)d_in[7];
    const float* ln1b  = (const float*)d_in[8];
    const float* W1    = (const float*)d_in[9];
    const float* W2    = (const float*)d_in[10];
    const float* ln2g  = (const float*)d_in[11];
    const float* ln2b  = (const float*)d_in[12];

    // ---- workspace layout ----
    char* ws = (char*)d_ws;
    const size_t nWqkvo = (size_t)L * D * D;
    const size_t nW1    = (size_t)L * D * DFF;
    const size_t nW2    = (size_t)L * DFF * D;

    _Float16* wq16 = (_Float16*)ws;        // all weights stored TRANSPOSED [N][K]
    _Float16* wk16 = wq16 + nWqkvo;
    _Float16* wv16 = wk16 + nWqkvo;
    _Float16* wo16 = wv16 + nWqkvo;
    _Float16* w116 = wo16 + nWqkvo;
    _Float16* w216 = w116 + nW1;
    char* p = (char*)(w216 + nW2);

    float*    x32   = (float*)p;              p += (size_t)S * D * 4;
    float*    tmp32 = (float*)p;              p += (size_t)S * D * 4;
    _Float16* x16   = (_Float16*)p;           p += (size_t)S * D * 2;
    _Float16* q16   = (_Float16*)p;           p += (size_t)S * D * 2;
    _Float16* k16   = (_Float16*)p;           p += (size_t)S * D * 2;
    _Float16* vt16  = (_Float16*)p;           p += (size_t)S * D * 2;
    _Float16* ctx16 = (_Float16*)p;           p += (size_t)S * D * 2;
    _Float16* h16   = (_Float16*)p;           p += (size_t)S * DFF * 2;
    float*    mvec  = (float*)p;              p += (size_t)S * 4;

    auto cgrid = [](size_t n) { return dim3((unsigned)((n + 255) / 256)); };
    // convert + transpose weights to f16 [N][K]
    convert_transpose_f16<<<cgrid(nWqkvo), 256, 0, stream>>>(Wq, wq16, D, D);
    convert_transpose_f16<<<cgrid(nWqkvo), 256, 0, stream>>>(Wk, wk16, D, D);
    convert_transpose_f16<<<cgrid(nWqkvo), 256, 0, stream>>>(Wv, wv16, D, D);
    convert_transpose_f16<<<cgrid(nWqkvo), 256, 0, stream>>>(Wo, wo16, D, D);
    convert_transpose_f16<<<cgrid(nW1),    256, 0, stream>>>(W1, w116, D, DFF);
    convert_transpose_f16<<<cgrid(nW2),    256, 0, stream>>>(W2, w216, DFF, D);

    pe_init_kernel<<<cgrid((size_t)S * D), 256, 0, stream>>>(enc, x32, x16);
    mvec_init_kernel<<<cgrid(S), 256, 0, stream>>>(mvec, Tp);
    mvec_global_kernel<<<1, 32, 0, stream>>>(mvec, gidx);

    const dim3 gD (D   / BN, S / BM);
    const dim3 gFF(DFF / BN, S / BM);
    const dim3 blk(256);
    const size_t sD = (size_t)D * D;

    for (int l = 0; l < L; ++l) {
        gemm_f16_wmma<<<gD, blk, 0, stream>>>(x16, wq16 + l * sD, nullptr, q16,  S, D, D, 0, 0);
        gemm_f16_wmma<<<gD, blk, 0, stream>>>(x16, wk16 + l * sD, nullptr, k16,  S, D, D, 0, 0);
        gemm_f16_wmma<<<gD, blk, 0, stream>>>(x16, wv16 + l * sD, nullptr, vt16, S, D, D, 1, S);

        attn_wmma<<<dim3(H * 16), blk, 0, stream>>>(q16, k16, vt16, mvec, ctx16);

        gemm_f16_wmma<<<gD, blk, 0, stream>>>(ctx16, wo16 + l * sD, x32, tmp32, S, D, D, 2, 0);
        layernorm_kernel<<<dim3(S), blk, 0, stream>>>(tmp32, ln1g + (size_t)l * D, ln1b + (size_t)l * D, x32, x16);

        gemm_f16_wmma<<<gFF, blk, 0, stream>>>(x16, w116 + (size_t)l * D * DFF, nullptr, h16, S, DFF, D, 3, 0);
        gemm_f16_wmma<<<gD,  blk, 0, stream>>>(h16, w216 + (size_t)l * DFF * D, x32, tmp32, S, D, DFF, 2, 0);
        layernorm_kernel<<<dim3(S), blk, 0, stream>>>(tmp32, ln2g + (size_t)l * D, ln2b + (size_t)l * D, x32, x16);
    }

    hipMemcpyAsync(d_out, x32, (size_t)S * D * sizeof(float), hipMemcpyDeviceToDevice, stream);
}
